// SelfAttentionBlock_53077205844465
// MI455X (gfx1250) — matrix operand order repaired and checked by reading, not compiled, not run on hardware
//
#include <hip/hip_runtime.h>
#include <hip/hip_bf16.h>

typedef __attribute__((ext_vector_type(16))) _Float16     v16h;
typedef __attribute__((ext_vector_type(8)))  float        v8f;
typedef __attribute__((ext_vector_type(8)))  unsigned int v8u;

#define Bb 16
#define Cc 512
#define Dd 64
#define Nn 4096

// ---------------------------------------------------------------------------
// WMMA helpers (wave32, gfx1250). A/B use contiguous k order (k = 16*(lane/16) + s) in both.
// ---------------------------------------------------------------------------
__device__ __forceinline__ v8f wmma_f16(v16h a, v16h b, v8f c) {
  return __builtin_amdgcn_wmma_f32_16x16x32_f16(
      /*neg_a=*/false, a, /*neg_b=*/false, b,
      /*c_mod=*/(short)0, c, /*reuse_a=*/false, /*reuse_b=*/false);
}

// A fragment: 16x32 f16, source LDS tile is row-major [row][k] with 'stride'.
// lanes 0-15: M=lane,  K=0..15
// lanes16-31: M=lane-16, K=16..31.
__device__ __forceinline__ v16h load_fragA(const _Float16* H, int stride,
                                           int rowBase, int kb, int lane) {
  const int g = lane >> 4;
  const int m = lane & 15;
  const _Float16* rp = H + (rowBase + m) * stride + kb + g * 16;
  v8u u;
#pragma unroll
  for (int v = 0; v < 8; ++v) {
    const int k = v << 1;
    u[v] = *(const unsigned int*)(rp + k);
  }
  return __builtin_bit_cast(v16h, u);
}

// B fragment: 32x16 f16 where B[k][n] = T[n][k] (T row-major, rows are n).
// lanes 0-15: N=lane, K=0..15; lanes 16-31: N=lane-16, K=16..31.
__device__ __forceinline__ v16h load_fragB_nk(const _Float16* T, int stride,
                                              int colBase, int kb, int lane) {
  const int g = lane >> 4;
  const int n = lane & 15;
  const _Float16* rp = T + (colBase + n) * stride + kb + g * 16;
  v8u u;
#pragma unroll
  for (int v = 0; v < 8; ++v) u[v] = *(const unsigned int*)(rp + v * 2);
  return __builtin_bit_cast(v16h, u);
}

// B fragment where LDS tile is stored [k][n] with 'stride'.
__device__ __forceinline__ v16h load_fragB_kn(const _Float16* T, int stride,
                                              int nBase, int kb, int lane) {
  const int g = lane >> 4;
  const int n = lane & 15;
  const _Float16* cp = T + (kb + g * 16) * stride + nBase + n;
  v16h r;
#pragma unroll
  for (int e = 0; e < 16; ++e) r[e] = cp[e * stride];
  return r;
}

__device__ __forceinline__ void split_store(float f, _Float16* H, _Float16* L, int o) {
  const _Float16 h = (_Float16)f;
  H[o] = h;
  L[o] = (_Float16)(f - (float)h);
}

// ---------------------------------------------------------------------------
// Kernel 1: G[b] = x[b] * x[b]^T   (512x512 per batch), plus sx row sums.
// Grid: (8 i-tiles, 8 j-tiles, 16 batches). Block 256 = 8 waves.
// ---------------------------------------------------------------------------
__global__ __launch_bounds__(256) void gram_kernel(const float* __restrict__ x,
                                                   float* __restrict__ G,
                                                   float* __restrict__ sx) {
  constexpr int STR = 72;
  const int ib = blockIdx.x, jb = blockIdx.y, b = blockIdx.z;
  const float* xb = x + (size_t)b * Cc * Nn;
  __shared__ _Float16 AH[64 * STR], AL[64 * STR], BH[64 * STR], BL[64 * STR];
  __shared__ float red[256];
  const int t = threadIdx.x;
  const int lane = t & 31, wave = t >> 5;
  const int mt  = (wave & 3) * 16;        // m tile base within 64
  const int nt0 = (wave >> 2) * 32;       // n base: 0 or 32 (two 16-tiles)
  const int lrow = t >> 2;                // load row 0..63
  const int lcol = (t & 3) * 16;          // load col base
  const bool diag = (ib == jb);
  v8f acc0 = {}, acc1 = {};
  float rsum = 0.f;

  for (int k0 = 0; k0 < Nn; k0 += 64) {
    __syncthreads();
    {
      const float* srcp = xb + (size_t)(ib * 64 + lrow) * Nn + k0 + lcol;
      if (k0 + 64 < Nn) __builtin_prefetch(srcp + 64, 0, 3);
      const float4* src = (const float4*)srcp;
#pragma unroll
      for (int q = 0; q < 4; ++q) {
        const float4 f = src[q];
        const int o = lrow * STR + lcol + q * 4;
        split_store(f.x, AH, AL, o + 0);
        split_store(f.y, AH, AL, o + 1);
        split_store(f.z, AH, AL, o + 2);
        split_store(f.w, AH, AL, o + 3);
        rsum += f.x + f.y + f.z + f.w;
      }
    }
    if (!diag) {
      const float4* src = (const float4*)(xb + (size_t)(jb * 64 + lrow) * Nn + k0 + lcol);
#pragma unroll
      for (int q = 0; q < 4; ++q) {
        const float4 f = src[q];
        const int o = lrow * STR + lcol + q * 4;
        split_store(f.x, BH, BL, o + 0);
        split_store(f.y, BH, BL, o + 1);
        split_store(f.z, BH, BL, o + 2);
        split_store(f.w, BH, BL, o + 3);
      }
    }
    __syncthreads();
    const _Float16* bh = diag ? AH : BH;
    const _Float16* bl = diag ? AL : BL;
#pragma unroll
    for (int kk = 0; kk < 64; kk += 32) {
      const v16h aH = load_fragA(AH, STR, mt, kk, lane);
      const v16h aL = load_fragA(AL, STR, mt, kk, lane);
      {
        const v16h bHf = load_fragB_nk(bh, STR, nt0, kk, lane);
        const v16h bLf = load_fragB_nk(bl, STR, nt0, kk, lane);
        acc0 = wmma_f16(aH, bHf, acc0);
        acc0 = wmma_f16(aH, bLf, acc0);
        acc0 = wmma_f16(aL, bHf, acc0);
      }
      {
        const v16h bHf = load_fragB_nk(bh, STR, nt0 + 16, kk, lane);
        const v16h bLf = load_fragB_nk(bl, STR, nt0 + 16, kk, lane);
        acc1 = wmma_f16(aH, bHf, acc1);
        acc1 = wmma_f16(aH, bLf, acc1);
        acc1 = wmma_f16(aL, bHf, acc1);
      }
    }
  }
  // store C tiles (f32 16x16 layout: lane<16 -> M=r, lane>=16 -> M=r+8, N=lane&15)
  {
    const int n = lane & 15, half = lane >> 4;
    float* dst = G + (size_t)b * Cc * Cc + (size_t)(ib * 64 + mt + half * 8) * Cc + jb * 64;
#pragma unroll
    for (int r = 0; r < 8; ++r) {
      dst[(size_t)r * Cc + nt0 + n]      = acc0[r];
      dst[(size_t)r * Cc + nt0 + 16 + n] = acc1[r];
    }
  }
  if (jb == 0) {   // fold row-sum computation into column-block 0
    red[t] = rsum;
    __syncthreads();
    if (t < 64) {
      const float s = red[t * 4] + red[t * 4 + 1] + red[t * 4 + 2] + red[t * 4 + 3];
      sx[b * Cc + ib * 64 + t] = s;
    }
  }
}

// ---------------------------------------------------------------------------
// Kernel 2a: P[b] = G[b] * Wk^T   ([512,64] per batch). Scalar fp32 (tiny).
// Grid: (8 c-tiles, 16 batches).
// ---------------------------------------------------------------------------
__global__ __launch_bounds__(256) void pk_kernel(const float* __restrict__ G,
                                                 const float* __restrict__ Wk,
                                                 float* __restrict__ P) {
  const int cb = blockIdx.x, b = blockIdx.y;
  const int c0 = cb * 64;
  const float* Gb = G + (size_t)b * Cc * Cc;
  __shared__ float Gt[64][65];
  __shared__ float Kt[64][65];
  const int t = threadIdx.x;
  const int lr = t >> 2, lc = (t & 3) * 16;
  const int cl = t >> 2, d0 = (t & 3) * 16;
  float acc[16];
#pragma unroll
  for (int i = 0; i < 16; ++i) acc[i] = 0.f;

  for (int j0 = 0; j0 < Cc; j0 += 64) {
    __syncthreads();
    const float4* gs = (const float4*)(Gb + (size_t)(c0 + lr) * Cc + j0 + lc);
    const float4* ks = (const float4*)(Wk + (size_t)lr * Cc + j0 + lc);
#pragma unroll
    for (int q = 0; q < 4; ++q) {
      const float4 f = gs[q];
      Gt[lr][lc + q * 4 + 0] = f.x; Gt[lr][lc + q * 4 + 1] = f.y;
      Gt[lr][lc + q * 4 + 2] = f.z; Gt[lr][lc + q * 4 + 3] = f.w;
      const float4 k = ks[q];
      Kt[lr][lc + q * 4 + 0] = k.x; Kt[lr][lc + q * 4 + 1] = k.y;
      Kt[lr][lc + q * 4 + 2] = k.z; Kt[lr][lc + q * 4 + 3] = k.w;
    }
    __syncthreads();
    for (int j = 0; j < 64; ++j) {
      const float gv = Gt[cl][j];
#pragma unroll
      for (int dd = 0; dd < 16; ++dd) acc[dd] += gv * Kt[d0 + dd][j];
    }
  }
  float* Pp = P + (size_t)b * Cc * Dd + (size_t)(c0 + cl) * Dd + d0;
#pragma unroll
  for (int dd = 0; dd < 16; ++dd) Pp[dd] = acc[dd];
}

// ---------------------------------------------------------------------------
// Kernel 2b: vk = Wv*P + u*bk^T + bv*(w1 + N*bk)^T ;  M = vk*Wq ; r = vk*bq
// Grid: (8 c-tiles, 16 batches). Scalar fp32 (tiny).
// ---------------------------------------------------------------------------
__global__ __launch_bounds__(256) void vkm_kernel(const float* __restrict__ P,
                                                  const float* __restrict__ Wv,
                                                  const float* __restrict__ Wk,
                                                  const float* __restrict__ Wq,
                                                  const float* __restrict__ bq,
                                                  const float* __restrict__ bk,
                                                  const float* __restrict__ bv,
                                                  const float* __restrict__ sx,
                                                  float* __restrict__ M,
                                                  float* __restrict__ rv) {
  const int cb = blockIdx.x, b = blockIdx.y;
  const int c0 = cb * 64;
  __shared__ float sxl[Cc];
  __shared__ float ul[64], w1l[64];
  __shared__ float vkl[64][65];
  __shared__ float TA[64][65];
  __shared__ float TB[64][65];
  const int t = threadIdx.x;
  const int cl = t >> 2, d0 = (t & 3) * 16;
  const int lr = t >> 2, lc = (t & 3) * 16;
  const float* Pb = P + (size_t)b * Cc * Dd;

  sxl[t]       = sx[b * Cc + t];
  sxl[t + 256] = sx[b * Cc + t + 256];
  __syncthreads();
  if (t < 64) {                       // u = Wv * sx (rows of this tile)
    float s = 0.f;
    const float* wr = Wv + (size_t)(c0 + t) * Cc;
    for (int j = 0; j < Cc; ++j) s += wr[j] * sxl[j];
    ul[t] = s;
  } else if (t < 128) {               // w1 = Wk * sx
    const int d = t - 64;
    float s = 0.f;
    const float* wr = Wk + (size_t)d * Cc;
    for (int j = 0; j < Cc; ++j) s += wr[j] * sxl[j];
    w1l[d] = s;
  }

  float acc[16];
#pragma unroll
  for (int i = 0; i < 16; ++i) acc[i] = 0.f;
  for (int j0 = 0; j0 < Cc; j0 += 64) {
    __syncthreads();
    const float4* as = (const float4*)(Wv + (size_t)(c0 + lr) * Cc + j0 + lc);
    const float4* bs = (const float4*)(Pb + (size_t)(j0 + lr) * Dd + lc);
#pragma unroll
    for (int q = 0; q < 4; ++q) {
      const float4 fa = as[q];
      TA[lr][lc + q * 4 + 0] = fa.x; TA[lr][lc + q * 4 + 1] = fa.y;
      TA[lr][lc + q * 4 + 2] = fa.z; TA[lr][lc + q * 4 + 3] = fa.w;
      const float4 fb = bs[q];
      TB[lr][lc + q * 4 + 0] = fb.x; TB[lr][lc + q * 4 + 1] = fb.y;
      TB[lr][lc + q * 4 + 2] = fb.z; TB[lr][lc + q * 4 + 3] = fb.w;
    }
    __syncthreads();
    for (int j = 0; j < 64; ++j) {
      const float a = TA[cl][j];
#pragma unroll
      for (int dd = 0; dd < 16; ++dd) acc[dd] += a * TB[j][d0 + dd];
    }
  }
  __syncthreads();
  {
    const float bvc = bv[c0 + cl];
    const float uc  = ul[cl];
#pragma unroll
    for (int dd = 0; dd < 16; ++dd) {
      const float bkd = bk[d0 + dd];
      vkl[cl][d0 + dd] = acc[dd] + uc * bkd + bvc * (w1l[d0 + dd] + (float)Nn * bkd);
    }
  }
  __syncthreads();

  for (int e0 = 0; e0 < Cc; e0 += 64) {   // M = vk * Wq
    __syncthreads();
    const float4* qs = (const float4*)(Wq + (size_t)lr * Cc + e0 + lc);
#pragma unroll
    for (int q = 0; q < 4; ++q) {
      const float4 f = qs[q];
      TA[lr][lc + q * 4 + 0] = f.x; TA[lr][lc + q * 4 + 1] = f.y;
      TA[lr][lc + q * 4 + 2] = f.z; TA[lr][lc + q * 4 + 3] = f.w;
    }
    __syncthreads();
    float m[16];
#pragma unroll
    for (int i = 0; i < 16; ++i) m[i] = 0.f;
    for (int d = 0; d < Dd; ++d) {
      const float v = vkl[cl][d];
#pragma unroll
      for (int ee = 0; ee < 16; ++ee) m[ee] += v * TA[d][d0 + ee];
    }
    float* Mp = M + (size_t)b * Cc * Cc + (size_t)(c0 + cl) * Cc + e0 + d0;
#pragma unroll
    for (int ee = 0; ee < 16; ++ee) Mp[ee] = m[ee];
  }

  if (t < 64) {                       // r = vk * bq
    float s = 0.f;
    for (int d = 0; d < Dd; ++d) s += vkl[t][d] * bq[d];
    rv[b * Cc + c0 + t] = s;
  }
}

// ---------------------------------------------------------------------------
// Kernel 3: out = gamma * (M * x + r*1^T) + x.  Tile 64(c) x 128(n) per block.
// Grid: (32 n-tiles, 8 c-tiles, 16 batches). Block 256 = 8 waves x 4 tiles.
// ---------------------------------------------------------------------------
__global__ __launch_bounds__(256) void out_kernel(const float* __restrict__ x,
                                                  const float* __restrict__ M,
                                                  const float* __restrict__ rv,
                                                  const float* __restrict__ gamma,
                                                  float* __restrict__ out) {
  constexpr int ASTR = 72, XSTR = 136;
  const int nb = blockIdx.x, cb = blockIdx.y, b = blockIdx.z;
  const int n0 = nb * 128, c0 = cb * 64;
  const float* xb = x + (size_t)b * Cc * Nn;
  const float* Mb = M + (size_t)b * Cc * Cc;
  __shared__ _Float16 AH[64 * ASTR], AL[64 * ASTR];
  __shared__ _Float16 XH[64 * XSTR], XL[64 * XSTR];
  const int t = threadIdx.x, lane = t & 31, wave = t >> 5;
  const int mt  = (wave & 3) * 16;
  const int ntb = (wave >> 2) * 64;       // 0 or 64
  const int lrow = t >> 2, lcol = (t & 3) * 16;
  const int xrow = t & 63, xcb = (t >> 6) * 32;
  v8f acc[4] = {};

  for (int k0 = 0; k0 < Cc; k0 += 64) {
    __syncthreads();
    {
      const float4* src = (const float4*)(Mb + (size_t)(c0 + lrow) * Cc + k0 + lcol);
#pragma unroll
      for (int q = 0; q < 4; ++q) {
        const float4 f = src[q];
        const int o = lrow * ASTR + lcol + q * 4;
        split_store(f.x, AH, AL, o + 0);
        split_store(f.y, AH, AL, o + 1);
        split_store(f.z, AH, AL, o + 2);
        split_store(f.w, AH, AL, o + 3);
      }
    }
    {
      const float* srcp = xb + (size_t)(k0 + xrow) * Nn + n0 + xcb;
      if (k0 + 64 < Cc) __builtin_prefetch(srcp + 64 * Nn, 0, 3);
      const float4* src = (const float4*)srcp;
#pragma unroll
      for (int q = 0; q < 8; ++q) {
        const float4 f = src[q];
        const int o = xrow * XSTR + xcb + q * 4;
        split_store(f.x, XH, XL, o + 0);
        split_store(f.y, XH, XL, o + 1);
        split_store(f.z, XH, XL, o + 2);
        split_store(f.w, XH, XL, o + 3);
      }
    }
    __syncthreads();
#pragma unroll
    for (int kk = 0; kk < 64; kk += 32) {
      const v16h aH = load_fragA(AH, ASTR, mt, kk, lane);
      const v16h aL = load_fragA(AL, ASTR, mt, kk, lane);
#pragma unroll
      for (int j = 0; j < 4; ++j) {
        const v16h bH = load_fragB_kn(XH, XSTR, ntb + j * 16, kk, lane);
        const v16h bL = load_fragB_kn(XL, XSTR, ntb + j * 16, kk, lane);
        acc[j] = wmma_f16(aH, bH, acc[j]);
        acc[j] = wmma_f16(aH, bL, acc[j]);
        acc[j] = wmma_f16(aL, bH, acc[j]);
      }
    }
  }
  const float g = gamma[0];
  const int n = lane & 15, half = lane >> 4;
#pragma unroll
  for (int r = 0; r < 8; ++r) {
    const int row = c0 + mt + half * 8 + r;
    const float rr = rv[b * Cc + row];
    const float* xr = xb + (size_t)row * Nn + n0;
    float* orow = out + (size_t)b * Cc * Nn + (size_t)row * Nn + n0;
#pragma unroll
    for (int j = 0; j < 4; ++j) {
      const int col = ntb + j * 16 + n;
      orow[col] = g * (acc[j][r] + rr) + xr[col];
    }
  }
}

// ---------------------------------------------------------------------------
extern "C" void kernel_launch(void* const* d_in, const int* in_sizes, int n_in,
                              void* d_out, int out_size, void* d_ws, size_t ws_size,
                              hipStream_t stream) {
  (void)in_sizes; (void)n_in; (void)out_size; (void)ws_size;
  const float* x     = (const float*)d_in[0];
  const float* Wq    = (const float*)d_in[1];
  const float* bq    = (const float*)d_in[2];
  const float* Wk    = (const float*)d_in[3];
  const float* bk    = (const float*)d_in[4];
  const float* Wv    = (const float*)d_in[5];
  const float* bv    = (const float*)d_in[6];
  const float* gamma = (const float*)d_in[7];
  float* out = (float*)d_out;

  // Workspace layout (floats): G[16*512*512] | sx[16*512] | P[16*512*64]
  //                          | M[16*512*512] | r[16*512]   (~35.7 MB total)
  float* ws = (float*)d_ws;
  float* G  = ws;
  float* sx = G  + (size_t)Bb * Cc * Cc;
  float* P  = sx + (size_t)Bb * Cc;
  float* M  = P  + (size_t)Bb * Cc * Dd;
  float* rv = M  + (size_t)Bb * Cc * Cc;

  gram_kernel<<<dim3(8, 8, Bb), 256, 0, stream>>>(x, G, sx);
  pk_kernel  <<<dim3(8, Bb),    256, 0, stream>>>(G, Wk, P);
  vkm_kernel <<<dim3(8, Bb),    256, 0, stream>>>(P, Wv, Wk, Wq, bq, bk, bv, sx, M, rv);
  out_kernel <<<dim3(32, 8, Bb), 256, 0, stream>>>(x, M, rv, gamma, out);
}
